// VarAutoEncoder_45054206935293
// MI455X (gfx1250) — compile-verified, matrix operand order checked by reading, not run on hardware
//
#include <hip/hip_runtime.h>
#include <hip/hip_bf16.h>
#include <math.h>

// ---------------- constants ----------------
#define NMOV   12288
#define LW     15
#define VOC    26
#define HID    100
#define CSZ    300
#define NL     (NMOV * LW)      // 184320
#define M3     (NMOV / 3)       // 4096

typedef __attribute__((ext_vector_type(16))) __bf16 v16bf;
typedef __attribute__((ext_vector_type(8)))  __bf16 v8bf;
typedef __attribute__((ext_vector_type(8)))  float  v8f;

// =========================================================================
// Generic GEMM:  C[M,N] = act( A[M,K] * B + bias )
//   A row-major, stride lda (MUST be even: float2 fast path).
//   transB==0 : B is [K,N] row-major;  transB==1 : B is [N,K] -> A*B^T
//   act: 0 = none, 1 = leaky_relu(0.01)
// Block: 128 thr = 4 waves. Block tile 64x32; wave w -> rows [16w,16w+16),
// two 16x16 WMMA tiles (cols 0-15 / 16-31) sharing the A fragment.
// LDS rows padded to 40 bf16 (80 B) so all fragment accesses are b128.
// B stored TRANSPOSED in LDS (Bt[n][k]) so B fragments match A's K-layout:
//   elems 0..7  -> K = 8g + 0..7      (one 16B load)
//   elems 8..15 -> K = 16 + 8g + 0..7 (one 16B load)
// =========================================================================
__global__ __launch_bounds__(128) void gemm_bf16_wmma(
    const float* __restrict__ A, int lda,
    const float* __restrict__ B, int ldb,
    const float* __restrict__ bias,
    float* __restrict__ C, int ldc,
    int M, int N, int K, int transB, int act)
{
    __shared__ alignas(16) __bf16 As[64][40];
    __shared__ alignas(16) __bf16 Bt[32][40];

    const int tid  = threadIdx.x;
    const int lane = tid & 31;
    const int wave = tid >> 5;           // 0..3 -> row strip
    const int row0 = blockIdx.x * 64;
    const int col0 = blockIdx.y * 32;
    const int g    = lane >> 4;          // lane half (K-phase select)
    const int nl   = lane & 15;

    v8f acc0 = {};
    v8f acc1 = {};

    for (int k0 = 0; k0 < K; k0 += 32) {
        // ---------------- stage A tile (64 x 32) ----------------
        if (row0 + 64 <= M && k0 + 32 <= K) {
            // fast path: thread t -> row t>>1, 16 contiguous cols, float2 loads
            int r   = tid >> 1;
            int seg = (tid & 1) * 16;
            const float* src = A + (size_t)(row0 + r) * lda + k0 + seg;
            float2 f[8];
            #pragma unroll
            for (int j = 0; j < 8; ++j)
                f[j] = *(const float2*)(src + 2 * j);
            v8bf lo, hi;
            #pragma unroll
            for (int j = 0; j < 4; ++j) {
                lo[2 * j]     = (__bf16)f[j].x;
                lo[2 * j + 1] = (__bf16)f[j].y;
                hi[2 * j]     = (__bf16)f[j + 4].x;
                hi[2 * j + 1] = (__bf16)f[j + 4].y;
            }
            *(v8bf*)&As[r][seg]     = lo;
            *(v8bf*)&As[r][seg + 8] = hi;
        } else {
            // guarded path: 16 batched scalar loads per thread (tails only)
            float tmp[16];
            #pragma unroll
            for (int j = 0; j < 16; ++j) {
                int i = tid + j * 128;
                int r = i >> 5, c = i & 31;
                int gr = row0 + r, gk = k0 + c;
                tmp[j] = (gr < M && gk < K) ? A[(size_t)gr * lda + gk] : 0.0f;
            }
            #pragma unroll
            for (int j = 0; j < 16; ++j) {
                int i = tid + j * 128;
                As[i >> 5][i & 31] = (__bf16)tmp[j];
            }
        }

        // ---------------- stage B tile transposed: Bt[n][k] ----------------
        {
            int n   = tid >> 2;            // 0..31
            int seg = (tid & 3) * 8;       // k sub-run
            int gn  = col0 + n;
            float tb[8];
            if (col0 + 32 <= N && k0 + 32 <= K) {
                // fast path: no guards, uniform branch on transB.
                // transB: contiguous k-run -> wide load clause.
                // !transB: clause of 8 strided b32 loads, single wait.
                const float* src = transB
                    ? (B + (size_t)gn * ldb + (k0 + seg))
                    : (B + (size_t)(k0 + seg) * ldb + gn);
                const size_t stride = transB ? 1 : (size_t)ldb;
                #pragma unroll
                for (int j = 0; j < 8; ++j) tb[j] = src[j * stride];
            } else {
                #pragma unroll
                for (int j = 0; j < 8; ++j) {
                    int gk = k0 + seg + j;
                    bool ok = (gn < N) && (gk < K);
                    size_t idx = transB ? (size_t)gn * ldb + gk
                                        : (size_t)gk * ldb + gn;
                    tb[j] = ok ? B[idx] : 0.0f;
                }
            }
            v8bf bb;
            #pragma unroll
            for (int j = 0; j < 8; ++j) bb[j] = (__bf16)tb[j];
            *(v8bf*)&Bt[n][seg] = bb;
        }
        __syncthreads();

        // ---------------- fragments: 2x b128 each ----------------
        const int mrow = wave * 16 + nl;
        v8bf alo = *(const v8bf*)&As[mrow][g * 8];
        v8bf ahi = *(const v8bf*)&As[mrow][16 + g * 8];
        v16bf af = __builtin_shufflevector(alo, ahi,
                     0,1,2,3,4,5,6,7,8,9,10,11,12,13,14,15);

        v8bf b0lo = *(const v8bf*)&Bt[nl][g * 8];
        v8bf b0hi = *(const v8bf*)&Bt[nl][16 + g * 8];
        v16bf bf0 = __builtin_shufflevector(b0lo, b0hi,
                     0,1,2,3,4,5,6,7,8,9,10,11,12,13,14,15);

        v8bf b1lo = *(const v8bf*)&Bt[16 + nl][g * 8];
        v8bf b1hi = *(const v8bf*)&Bt[16 + nl][16 + g * 8];
        v16bf bf1 = __builtin_shufflevector(b1lo, b1hi,
                     0,1,2,3,4,5,6,7,8,9,10,11,12,13,14,15);

        acc0 = __builtin_amdgcn_wmma_f32_16x16x32_bf16(
                   false, af, false, bf0, (short)0, acc0, false, false);
        acc1 = __builtin_amdgcn_wmma_f32_16x16x32_bf16(
                   false, af, false, bf1, (short)0, acc1, false, false);
        __syncthreads();
    }

    // ---------------- epilogue ----------------
    #pragma unroll
    for (int t = 0; t < 2; ++t) {
        const v8f& a = t ? acc1 : acc0;
        int gc = col0 + t * 16 + nl;
        if (gc < N) {
            float bv = bias ? bias[gc] : 0.0f;
            #pragma unroll
            for (int j = 0; j < 8; ++j) {
                int gr = row0 + wave * 16 + g * 8 + j;
                if (gr < M) {
                    float v = a[j] + bv;
                    if (act == 1) v = (v > 0.0f) ? v : 0.01f * v;
                    C[(size_t)gr * ldc + gc] = v;
                }
            }
        }
    }
}

// ---------------- embedding gather: emb[n,l,:] = letter_emb[tok]+pos_emb[l] --
__global__ void embed_kernel(const int* __restrict__ tok,
                             const float* __restrict__ letter_emb,
                             const float* __restrict__ pos_emb,
                             float* __restrict__ emb)
{
    size_t idx = (size_t)blockIdx.x * blockDim.x + threadIdx.x;
    if (idx >= (size_t)NL * HID) return;
    int nl = (int)(idx / HID);
    int h  = (int)(idx % HID);
    int l  = nl % LW;
    int t  = tok[nl];
    emb[idx] = letter_emb[t * HID + h] + pos_emb[l * HID + h];
}

// ---------------- per-move 15x15 self-attention, sum over L, tanh -----------
// one wave per move; colsum trick: sum_l attn[l,:] = sum_m (sum_l P[l,m]) v[m,:]
__global__ __launch_bounds__(128) void enc_attn_kernel(
    const float* __restrict__ q, const float* __restrict__ k,
    const float* __restrict__ v, float* __restrict__ word_enc)
{
    __shared__ float S[4][LW][LW + 1];
    __shared__ float colsum[4][LW + 1];
    const int wave = threadIdx.x >> 5, lane = threadIdx.x & 31;
    const int move = blockIdx.x * 4 + wave;
    const bool act = move < NMOV;
    const float* qm = q + (size_t)move * LW * HID;
    const float* km = k + (size_t)move * LW * HID;
    const float* vm = v + (size_t)move * LW * HID;

    if (act) {
        for (int p = lane; p < LW * LW; p += 32) {
            int l = p / LW, m = p % LW;
            float d = 0.0f;
            for (int h = 0; h < HID; ++h) d += qm[l * HID + h] * km[m * HID + h];
            S[wave][l][m] = d;
        }
    }
    __syncthreads();
    if (act && lane < LW) {
        int l = lane;
        float mx = -3.4e38f;
        for (int m = 0; m < LW; ++m) mx = fmaxf(mx, S[wave][l][m]);
        float sum = 0.0f;
        for (int m = 0; m < LW; ++m) { float e = expf(S[wave][l][m] - mx); S[wave][l][m] = e; sum += e; }
        float inv = 1.0f / sum;
        for (int m = 0; m < LW; ++m) S[wave][l][m] *= inv;
    }
    __syncthreads();
    if (act && lane < LW) {
        int m = lane; float cs = 0.0f;
        for (int l = 0; l < LW; ++l) cs += S[wave][l][m];
        colsum[wave][m] = cs;
    }
    __syncthreads();
    if (act) {
        for (int h = lane; h < HID; h += 32) {
            float a = 0.0f;
            for (int m = 0; m < LW; ++m) a += colsum[wave][m] * vm[m * HID + h];
            word_enc[(size_t)move * HID + h] = tanhf(a);
        }
    }
}

// ---------------- score min/max via uint-bit atomics (scores >= 0) ----------
__global__ void minmax_init_kernel(unsigned* mm)
{
    if (threadIdx.x == 0 && blockIdx.x == 0) { mm[0] = 0x7f7fffffu; mm[1] = 0u; }
}
__global__ void minmax_kernel(const float* __restrict__ s, unsigned* mm)
{
    int i = blockIdx.x * blockDim.x + threadIdx.x;
    if (i < NMOV) {
        unsigned b = __float_as_uint(s[i]);
        atomicMin(&mm[0], b);
        atomicMax(&mm[1], b);
    }
}

// ---------------- build x = [we,we,vec_enc] + s*score_w + score_b -----------
__global__ void build_x_kernel(
    const float* __restrict__ word_enc,
    const int* __restrict__ rows, const int* __restrict__ cols,
    const int* __restrict__ dirs,
    const float* __restrict__ row_emb, const float* __restrict__ col_emb,
    const float* __restrict__ dir_emb,
    const float* __restrict__ scores, const unsigned* __restrict__ mm,
    const float* __restrict__ score_w, const float* __restrict__ score_b,
    float* __restrict__ x)
{
    size_t idx = (size_t)blockIdx.x * blockDim.x + threadIdx.x;
    if (idx >= (size_t)NMOV * CSZ) return;
    int n = (int)(idx / CSZ);
    int c = (int)(idx % CSZ);
    float mn = __uint_as_float(mm[0]);
    float mx = __uint_as_float(mm[1]);
    float d  = mx - mn;
    float sn = (scores[n] - mn) / ((d == 0.0f) ? 1.0f : d);
    float val;
    if (c < 200) {
        val = word_enc[(size_t)n * HID + (c % HID)];
    } else {
        int h = c - 200;
        val = tanhf(row_emb[rows[n] * HID + h] + col_emb[cols[n] * HID + h] +
                    dir_emb[dirs[n] * HID + h]);
    }
    x[idx] = val + sn * score_w[c] + score_b[c];
}

// ---------------- encoded = mu + exp(0.5*lv)*eps ----------------------------
__global__ void reparam_kernel(const float* __restrict__ mu,
                               const float* __restrict__ lv,
                               const float* __restrict__ eps,
                               float* __restrict__ enc)
{
    size_t i = (size_t)blockIdx.x * blockDim.x + threadIdx.x;
    if (i < (size_t)NMOV * HID) enc[i] = mu[i] + expf(0.5f * lv[i]) * eps[i];
}

// ---------------- elementwise a - b -----------------------------------------
__global__ void sub_kernel(const float* __restrict__ a, const float* __restrict__ b,
                           float* __restrict__ o, size_t n)
{
    size_t i = (size_t)blockIdx.x * blockDim.x + threadIdx.x;
    if (i < n) o[i] = a[i] - b[i];
}

// ---------------- score head: out[n] = dot(sb2[n,:], w) + b -----------------
__global__ void score_dec_kernel(const float* __restrict__ sb2,
                                 const float* __restrict__ w,
                                 const float* __restrict__ b,
                                 float* __restrict__ out)
{
    int n = blockIdx.x * blockDim.x + threadIdx.x;
    if (n >= NMOV) return;
    float s = 0.0f;
    for (int h = 0; h < HID; ++h) s += sb2[(size_t)n * HID + h] * w[h];
    out[n] = s + b[0];
}

// ---------------- row softmax, small cols (<= 32), one thread per row -------
__global__ void softmax_small_kernel(const float* __restrict__ in,
                                     float* __restrict__ out,
                                     int rows, int cols)
{
    int r = blockIdx.x * blockDim.x + threadIdx.x;
    if (r >= rows) return;
    const float* p = in + (size_t)r * cols;
    float* o = out + (size_t)r * cols;
    float e[32];
    float mx = -3.4e38f;
    for (int c = 0; c < cols; ++c) mx = fmaxf(mx, p[c]);
    float sum = 0.0f;
    for (int c = 0; c < cols; ++c) { e[c] = expf(p[c] - mx); sum += e[c]; }
    float inv = 1.0f / sum;
    for (int c = 0; c < cols; ++c) o[c] = e[c] * inv;
}

// ---------------- row softmax in-place, wide rows (4096), block per row -----
__global__ __launch_bounds__(256) void softmax_big_kernel(float* __restrict__ S, int cols)
{
    __shared__ float red[256];
    int row = blockIdx.x, tid = threadIdx.x;
    float* p = S + (size_t)row * cols;
    float mx = -3.4e38f;
    for (int c = tid; c < cols; c += 256) mx = fmaxf(mx, p[c]);
    red[tid] = mx; __syncthreads();
    for (int s = 128; s > 0; s >>= 1) { if (tid < s) red[tid] = fmaxf(red[tid], red[tid + s]); __syncthreads(); }
    mx = red[0]; __syncthreads();
    float sum = 0.0f;
    for (int c = tid; c < cols; c += 256) { float e = expf(p[c] - mx); p[c] = e; sum += e; }
    red[tid] = sum; __syncthreads();
    for (int s = 128; s > 0; s >>= 1) { if (tid < s) red[tid] += red[tid + s]; __syncthreads(); }
    float inv = 1.0f / red[0];
    for (int c = tid; c < cols; c += 256) p[c] *= inv;
}

// =========================================================================
// Host orchestration
// =========================================================================
static inline void gemm(hipStream_t s, const float* A, int lda,
                        const float* B, int ldb, const float* bias,
                        float* C, int ldc, int M, int N, int K,
                        int transB, int act)
{
    dim3 grid((M + 63) / 64, (N + 31) / 32);
    gemm_bf16_wmma<<<grid, 128, 0, s>>>(A, lda, B, ldb, bias, C, ldc,
                                        M, N, K, transB, act);
}

extern "C" void kernel_launch(void* const* d_in, const int* in_sizes, int n_in,
                              void* d_out, int out_size, void* d_ws, size_t ws_size,
                              hipStream_t stream)
{
    // ---- inputs (setup_inputs order) ----
    const int*   word_tokens = (const int*)  d_in[0];
    const int*   rows        = (const int*)  d_in[1];
    const int*   cols        = (const int*)  d_in[2];
    const int*   dirs        = (const int*)  d_in[3];
    const float* scores_in   = (const float*)d_in[4];
    const float* epsilon     = (const float*)d_in[5];
    const float* letter_emb  = (const float*)d_in[6];
    const float* pos_emb     = (const float*)d_in[7];
    const float* enc_sa_wq   = (const float*)d_in[8];
    const float* enc_sa_bq   = (const float*)d_in[9];
    const float* enc_sa_wk   = (const float*)d_in[10];
    const float* enc_sa_bk   = (const float*)d_in[11];
    const float* enc_sa_wv   = (const float*)d_in[12];
    const float* enc_sa_bv   = (const float*)d_in[13];
    const float* row_emb     = (const float*)d_in[14];
    const float* col_emb     = (const float*)d_in[15];
    const float* dir_emb     = (const float*)d_in[16];
    const float* score_enc_w = (const float*)d_in[17];
    const float* score_enc_b = (const float*)d_in[18];
    const float* ds_w1 = (const float*)d_in[19]; const float* ds_b1 = (const float*)d_in[20];
    const float* ds_w2 = (const float*)d_in[21]; const float* ds_b2 = (const float*)d_in[22];
    const float* ds_w3 = (const float*)d_in[23]; const float* ds_b3 = (const float*)d_in[24];
    const float* mu_w  = (const float*)d_in[25]; const float* mu_b  = (const float*)d_in[26];
    const float* lv_w  = (const float*)d_in[27]; const float* lv_b  = (const float*)d_in[28];
    const float* up_w1 = (const float*)d_in[29]; const float* up_b1 = (const float*)d_in[30];
    const float* up_w2 = (const float*)d_in[31]; const float* up_b2 = (const float*)d_in[32];
    const float* up_w3 = (const float*)d_in[33]; const float* up_b3 = (const float*)d_in[34];
    const float* sep_w = (const float*)d_in[35]; const float* sep_b = (const float*)d_in[36];
    const float* sdec_w = (const float*)d_in[37]; const float* sdec_b = (const float*)d_in[38];
    const float* dec_sa_wq = (const float*)d_in[39]; const float* dec_sa_bq = (const float*)d_in[40];
    const float* dec_sa_wk = (const float*)d_in[41]; const float* dec_sa_bk = (const float*)d_in[42];
    const float* dec_sa_wv = (const float*)d_in[43]; const float* dec_sa_bv = (const float*)d_in[44];
    const float* wn_w1 = (const float*)d_in[45]; const float* wn_b1 = (const float*)d_in[46];
    const float* wn_w2 = (const float*)d_in[47]; const float* wn_b2 = (const float*)d_in[48];
    const float* wn_w3 = (const float*)d_in[49]; const float* wn_b3 = (const float*)d_in[50];
    const float* vrow_w = (const float*)d_in[51];
    const float* vcol_w = (const float*)d_in[52];
    const float* vdir_w = (const float*)d_in[53];
    const float* len_w1 = (const float*)d_in[54]; const float* len_b1 = (const float*)d_in[55];
    const float* len_w2 = (const float*)d_in[56]; const float* len_b2 = (const float*)d_in[57];
    const float* len_w3 = (const float*)d_in[58]; const float* len_b3 = (const float*)d_in[59];

    float* out = (float*)d_out;
    // output layout (flat concat, reference return order)
    const size_t len_off   = 0;
    const size_t word_off  = len_off  + (size_t)M3 * LW;
    const size_t ord_off   = word_off + (size_t)M3 * VOC;
    const size_t row_off   = ord_off  + (size_t)M3 * VOC;
    const size_t col_off   = row_off  + (size_t)M3 * LW;
    const size_t dir_off   = col_off  + (size_t)M3 * LW;
    const size_t score_off = dir_off  + (size_t)M3 * 2;
    const size_t mu_off    = score_off + (size_t)NMOV;
    const size_t lv_off    = mu_off   + (size_t)NMOV * HID;

    // ---- workspace bump allocator ----
    size_t woff = 0;
    auto alloc = [&](size_t nfloats) -> float* {
        float* p = (float*)((char*)d_ws + woff);
        woff += ((nfloats * sizeof(float) + 255) / 256) * 256;
        return p;
    };
    unsigned* mm  = (unsigned*)alloc(64);
    float* emb    = alloc((size_t)NL * HID);
    float* q      = alloc((size_t)NL * HID);
    float* k      = alloc((size_t)NL * HID);
    float* v      = alloc((size_t)NL * HID);
    float* wenc   = alloc((size_t)NMOV * HID);
    float* x      = alloc((size_t)NMOV * CSZ);
    float* h1     = alloc((size_t)NMOV * 200);
    float* h2     = alloc((size_t)NMOV * 150);
    float* down   = alloc((size_t)NMOV * HID);
    float* enc    = alloc((size_t)NMOV * HID);
    float* u1     = alloc((size_t)NMOV * 150);
    float* u2     = alloc((size_t)NMOV * 200);
    float* up     = alloc((size_t)NMOV * HID);
    float* sb2    = alloc((size_t)NMOV * HID);
    float* mdec   = alloc((size_t)NMOV * HID);
    float* l1     = alloc((size_t)M3 * 100);
    float* l2     = alloc((size_t)M3 * 50);
    float* l3     = alloc((size_t)M3 * LW);
    float* qq     = alloc((size_t)M3 * HID);
    float* kk     = alloc((size_t)M3 * HID);
    float* vv     = alloc((size_t)M3 * HID);
    float* S      = alloc((size_t)M3 * M3);
    float* attn   = alloc((size_t)M3 * HID);
    float* wh1    = alloc((size_t)M3 * 150);
    float* wh2    = alloc((size_t)M3 * 200);
    float* wlog   = alloc((size_t)M3 * VOC);
    (void)ws_size; (void)n_in; (void)in_sizes; (void)out_size;

    // ---- 1. encoder embeddings + QKV projections (WMMA GEMMs over NL rows)
    {
        size_t tot = (size_t)NL * HID;
        embed_kernel<<<(tot + 255) / 256, 256, 0, stream>>>(word_tokens, letter_emb, pos_emb, emb);
    }
    gemm(stream, emb, HID, enc_sa_wq, HID, enc_sa_bq, q, HID, NL, HID, HID, 0, 0);
    gemm(stream, emb, HID, enc_sa_wk, HID, enc_sa_bk, k, HID, NL, HID, HID, 0, 0);
    gemm(stream, emb, HID, enc_sa_wv, HID, enc_sa_bv, v, HID, NL, HID, HID, 0, 0);

    // ---- 2. per-move 15x15 attention -> word_enc
    enc_attn_kernel<<<NMOV / 4, 128, 0, stream>>>(q, k, v, wenc);

    // ---- 3. score normalization + x assembly
    minmax_init_kernel<<<1, 1, 0, stream>>>(mm);
    minmax_kernel<<<(NMOV + 255) / 256, 256, 0, stream>>>(scores_in, mm);
    {
        size_t tot = (size_t)NMOV * CSZ;
        build_x_kernel<<<(tot + 255) / 256, 256, 0, stream>>>(
            wenc, rows, cols, dirs, row_emb, col_emb, dir_emb,
            scores_in, mm, score_enc_w, score_enc_b, x);
    }

    // ---- 4. downsampler MLP + mu/lv (mu/lv written straight into d_out)
    gemm(stream, x,  CSZ, ds_w1, 200, ds_b1, h1,   200, NMOV, 200, CSZ, 0, 1);
    gemm(stream, h1, 200, ds_w2, 150, ds_b2, h2,   150, NMOV, 150, 200, 0, 1);
    gemm(stream, h2, 150, ds_w3, HID, ds_b3, down, HID, NMOV, HID, 150, 0, 0);
    gemm(stream, down, HID, mu_w, HID, mu_b, out + mu_off, HID, NMOV, HID, HID, 0, 0);
    gemm(stream, down, HID, lv_w, HID, lv_b, out + lv_off, HID, NMOV, HID, HID, 0, 0);
    {
        size_t tot = (size_t)NMOV * HID;
        reparam_kernel<<<(tot + 255) / 256, 256, 0, stream>>>(out + mu_off, out + lv_off, epsilon, enc);
    }

    // ---- 5. decoder upsampler + separation + score head
    gemm(stream, enc, HID, up_w1, 150, up_b1, u1, 150, NMOV, 150, HID, 0, 1);
    gemm(stream, u1, 150, up_w2, 200, up_b2, u2, 200, NMOV, 200, 150, 0, 1);
    gemm(stream, u2, 200, up_w3, HID, up_b3, up, HID, NMOV, HID, 200, 0, 0);
    gemm(stream, up, HID, sep_w, HID, sep_b, sb2, HID, NMOV, HID, HID, 0, 0);
    {
        size_t tot = (size_t)NMOV * HID;
        sub_kernel<<<(tot + 255) / 256, 256, 0, stream>>>(up, sb2, mdec, tot);
    }
    score_dec_kernel<<<(NMOV + 255) / 256, 256, 0, stream>>>(sb2, sdec_w, sdec_b, out + score_off);

    // mdec is [NMOV, HID]; we/oe/ve are lda=300 views: +0 / +100 / +200
    const float* we = mdec;              // rows 3i
    const float* oe = mdec + HID;        // rows 3i+1
    const float* ve = mdec + 2 * HID;    // rows 3i+2

    // ---- 6. length head: concat(we,oe) is contiguous 200 floats @ mdec+300i
    gemm(stream, mdec, 3 * HID, len_w1, 100, len_b1, l1, 100, M3, 100, 200, 0, 1);
    gemm(stream, l1, 100, len_w2, 50, len_b2, l2, 50, M3, 50, 100, 0, 1);
    gemm(stream, l2, 50, len_w3, LW, len_b3, l3, LW, M3, LW, 50, 0, 0);
    softmax_small_kernel<<<(M3 + 255) / 256, 256, 0, stream>>>(l3, out + len_off, M3, LW);

    // ---- 7. word decoder (run twice: we -> word_raw, oe -> ordered_raw)
    const float* xs[2]   = { we, oe };
    const size_t offs[2] = { word_off, ord_off };
    for (int pass = 0; pass < 2; ++pass) {
        const float* xx = xs[pass];
        gemm(stream, xx, 3 * HID, dec_sa_wq, HID, dec_sa_bq, qq, HID, M3, HID, HID, 0, 0);
        gemm(stream, xx, 3 * HID, dec_sa_wk, HID, dec_sa_bk, kk, HID, M3, HID, HID, 0, 0);
        gemm(stream, xx, 3 * HID, dec_sa_wv, HID, dec_sa_bv, vv, HID, M3, HID, HID, 0, 0);
        // S = qq * kk^T   [4096 x 4096]
        gemm(stream, qq, HID, kk, HID, nullptr, S, M3, M3, M3, HID, 1, 0);
        softmax_big_kernel<<<M3, 256, 0, stream>>>(S, M3);
        // attn = P * vv   [4096 x 100], K = 4096
        gemm(stream, S, M3, vv, HID, nullptr, attn, HID, M3, HID, M3, 0, 0);
        gemm(stream, attn, HID, wn_w1, 150, wn_b1, wh1, 150, M3, 150, HID, 0, 1);
        gemm(stream, wh1, 150, wn_w2, 200, wn_b2, wh2, 200, M3, 200, 150, 0, 1);
        gemm(stream, wh2, 200, wn_w3, VOC, wn_b3, wlog, VOC, M3, VOC, 200, 0, 0);
        softmax_small_kernel<<<(M3 + 255) / 256, 256, 0, stream>>>(wlog, out + offs[pass], M3, VOC);
    }

    // ---- 8. vector decoder heads (plain GEMMs, no bias) straight to d_out
    gemm(stream, ve, 3 * HID, vrow_w, LW, nullptr, out + row_off, LW, M3, LW, HID, 0, 0);
    gemm(stream, ve, 3 * HID, vcol_w, LW, nullptr, out + col_off, LW, M3, LW, HID, 0, 0);
    gemm(stream, ve, 3 * HID, vdir_w, 2,  nullptr, out + dir_off, 2,  M3, 2,  HID, 0, 0);
}
